// MultimerInvariantPointAttention_80092550136255
// MI455X (gfx1250) — matrix operand order repaired and checked by reading, not compiled, not run on hardware
//
#include <hip/hip_runtime.h>
#include <hip/hip_bf16.h>
#include <math.h>

// ---------------------------------------------------------------------------
// Multimer Invariant Point Attention for MI455X (gfx1250, wave32, WMMA).
// Contractions via v_wmma_f32_16x16x32_bf16 (f32 accum). Bandwidth-bound:
// ~320MB of traffic / 23.3 TB/s ~= 14us floor; inputs_2d (134MB) < L2 (192MB)
// so the second streaming pass is L2-resident. Streaming tiles of inputs_2d
// are staged with CDNA5 global_load_async_to_lds_b128 (ASYNCcnt path).
// ---------------------------------------------------------------------------

typedef __attribute__((ext_vector_type(16))) __bf16 v16bf;
typedef __attribute__((ext_vector_type(8)))  float  v8f;

#define N_RES   512
#define C_DIM   384
#define H_NUM   12
#define SQK_D   16
#define SV_D    16
#define PQK_D   4
#define PV_D    8
#define PAIR_D  128
#define ACT_DIM 2112   // H*(SV + 4*PV + PAIR) = 12*176

// --- CDNA5 async copy global->LDS (tracked by ASYNCcnt) --------------------
#define STR2(x) #x
#define ASYNC_B128(ldsv, gav, OFF)                                           \
  asm volatile("global_load_async_to_lds_b128 %0, %1, off offset:" STR2(OFF) \
               ::"v"(ldsv), "v"(gav) : "memory")
#define S_WAIT_ASYNC0 asm volatile("s_wait_asynccnt 0x0" ::: "memory")
#define S_WAIT_DS0    asm volatile("s_wait_dscnt 0x0" ::: "memory")

static __device__ __forceinline__ v8f wmma_bf16(v16bf a, v16bf b, v8f c) {
  return __builtin_amdgcn_wmma_f32_16x16x32_bf16(false, a, false, b, (short)0, c,
                                                 false, false);
}

// A-frag (16x32) from a row-major bf16 LDS tile (ISA 7.12.2 layout):
// lane row m=lane&15, K-base 0/8 by lane half, elem e -> K = kb + e + (e&8)
static __device__ __forceinline__ v16bf load_A_lds(const __bf16* t, int lda, int lane) {
  int m  = lane & 15;
  int kb = (lane & 16) ? 8 : 0;
  v16bf a;
#pragma unroll
  for (int e = 0; e < 16; ++e) a[e] = t[m * lda + kb + e + (e & 8)];
  return a;
}

// A-frag built from f32 memory (global or LDS): two contiguous 8-float runs
// -> 4 x b128 loads. `row` must already point at [... + k0 + kbA].
static __device__ __forceinline__ v16bf frag_A_f32(const float* row) {
  const float4* p = (const float4*)row;
  float4 f0 = p[0], f1 = p[1], f2 = p[4], f3 = p[5];
  v16bf a;
  a[0] = (__bf16)f0.x; a[1] = (__bf16)f0.y; a[2]  = (__bf16)f0.z; a[3]  = (__bf16)f0.w;
  a[4] = (__bf16)f1.x; a[5] = (__bf16)f1.y; a[6]  = (__bf16)f1.z; a[7]  = (__bf16)f1.w;
  a[8] = (__bf16)f2.x; a[9] = (__bf16)f2.y; a[10] = (__bf16)f2.z; a[11] = (__bf16)f2.w;
  a[12] = (__bf16)f3.x; a[13] = (__bf16)f3.y; a[14] = (__bf16)f3.z; a[15] = (__bf16)f3.w;
  return a;
}

// B-frag (32x16) from an LDS tile stored TRANSPOSED [N][K] so each lane's
// 16 bf16 are two contiguous 8-runs (2 x ds_load_b128). K = kb + e + (e&8), kb = 0/8 by lane half.
static __device__ __forceinline__ v16bf load_B_ldsT(const __bf16* t, int ldk,
                                                    int k0, int lane) {
  int n  = lane & 15;
  int kb = (lane & 16) ? 8 : 0;
  v16bf b;
#pragma unroll
  for (int e = 0; e < 16; ++e) b[e] = t[n * ldk + k0 + kb + e + (e & 8)];
  return b;
}

// ---------------------------------------------------------------------------
// Generic C = A(MxK,f32) @ B(KxN,f32) [+ bias(N)] via bf16 WMMA.
// One wave per 16x16 tile; M,N multiples of 16; K multiple of 32.
// ---------------------------------------------------------------------------
__global__ __launch_bounds__(32) void gemm_bf16_kernel(
    const float* __restrict__ A, const float* __restrict__ B,
    const float* __restrict__ bias, float* __restrict__ C,
    int M, int N, int K) {
  __shared__ __bf16 sA[16 * 32];
  __shared__ __bf16 sBt[16 * 32];   // transposed [n][k]
  int lane = threadIdx.x;
  int m0 = blockIdx.x * 16, n0 = blockIdx.y * 16;
  v8f acc = {};
  for (int k0 = 0; k0 < K; k0 += 32) {
    int r  = lane >> 1;
    int cb = (lane & 1) << 4;
    {  // A: 16 contiguous f32 per lane -> 4 x global_load_b128
      const float4* ap = (const float4*)(A + (size_t)(m0 + r) * K + k0 + cb);
      float4 f0 = ap[0], f1 = ap[1], f2 = ap[2], f3 = ap[3];
      __bf16* d = &sA[r * 32 + cb];
      d[0] = (__bf16)f0.x; d[1] = (__bf16)f0.y; d[2]  = (__bf16)f0.z; d[3]  = (__bf16)f0.w;
      d[4] = (__bf16)f1.x; d[5] = (__bf16)f1.y; d[6]  = (__bf16)f1.z; d[7]  = (__bf16)f1.w;
      d[8] = (__bf16)f2.x; d[9] = (__bf16)f2.y; d[10] = (__bf16)f2.z; d[11] = (__bf16)f2.w;
      d[12] = (__bf16)f3.x; d[13] = (__bf16)f3.y; d[14] = (__bf16)f3.z; d[15] = (__bf16)f3.w;
    }
    {  // B row k0+lane, 16 contiguous f32 -> 4 x b128; store transposed
      const float4* bp = (const float4*)(B + (size_t)(k0 + lane) * N + n0);
      float4 f0 = bp[0], f1 = bp[1], f2 = bp[2], f3 = bp[3];
      float v[16] = {f0.x, f0.y, f0.z, f0.w, f1.x, f1.y, f1.z, f1.w,
                     f2.x, f2.y, f2.z, f2.w, f3.x, f3.y, f3.z, f3.w};
#pragma unroll
      for (int i = 0; i < 16; ++i) sBt[i * 32 + lane] = (__bf16)v[i];
    }
    __syncthreads();
    acc = wmma_bf16(load_A_lds(sA, 32, lane), load_B_ldsT(sBt, 32, 0, lane), acc);
    __syncthreads();
  }
  int n  = n0 + (lane & 15);
  int mb = m0 + ((lane >> 4) << 3);
  float bv = bias ? bias[n] : 0.f;
#pragma unroll
  for (int r = 0; r < 8; ++r)
    C[(size_t)(mb + r) * N + n] = acc[r] + bv;
}

// ---------------------------------------------------------------------------
// Rotate+translate projected points into the global frame.
// raw: (N, H*3*P) row-major; out: (N, 3, H, P)
// ---------------------------------------------------------------------------
__global__ void to_global_kernel(const float* __restrict__ raw,
                                 const float* __restrict__ rot,
                                 const float* __restrict__ trans,
                                 float* __restrict__ out, int P) {
  int n = blockIdx.x;
  int HP = H_NUM * P;
  int idx = threadIdx.x;           // i*HP + h*P + p
  if (idx >= 3 * HP) return;
  int i = idx / HP;
  int hp = idx % HP;
  int h = hp / P, p = hp % P;
  const float* R = rot + n * 9;
  float s = trans[n * 3 + i];
#pragma unroll
  for (int j = 0; j < 3; ++j)
    s += R[i * 3 + j] * raw[(size_t)n * (3 * HP) + h * (3 * P) + j * P + p];
  out[((size_t)n * 3 + i) * HP + hp] = s;
}

// ---------------------------------------------------------------------------
// logits init: pair bias (inputs_2d @ W_attn2d + b) + mask term.
// M = flattened (q,k) pairs, K = 128, N = 12 (padded to 16).
// The 16x128 f32 A tile is a CONTIGUOUS 8KB block in global memory: copy it
// to LDS with 16 async b128 ops (global stride 512B == LDS stride 512B).
// ---------------------------------------------------------------------------
__global__ __launch_bounds__(32) void bias_logits_kernel(
    const float* __restrict__ in2d, const float* __restrict__ Wb,
    const float* __restrict__ bb, const float* __restrict__ mask,
    float* __restrict__ logits) {
  __shared__ float  sAf[16 * PAIR_D];     // 8KB raw f32 tile
  __shared__ __bf16 sBt[16 * PAIR_D];     // transposed [n][k] 4KB
  int lane = threadIdx.x;
  int pair0 = blockIdx.x * 16;

  // stage B once (128x12 weights, zero-padded to 16 cols), transposed
#pragma unroll
  for (int r4 = 0; r4 < 4; ++r4) {
    int row = lane + 32 * r4;
#pragma unroll
    for (int i = 0; i < 16; ++i) {
      float v = (i < H_NUM) ? Wb[row * H_NUM + i] : 0.f;
      sBt[i * PAIR_D + row] = (__bf16)v;
    }
  }

  // async copy of the contiguous 8KB A tile into LDS
  unsigned ldsa = (unsigned)(size_t)sAf + (unsigned)lane * 16u;
  unsigned long long ga =
      (unsigned long long)(size_t)(in2d + (size_t)pair0 * PAIR_D) +
      (unsigned long long)lane * 16ull;
  ASYNC_B128(ldsa, ga, 0);    ASYNC_B128(ldsa, ga, 512);
  ASYNC_B128(ldsa, ga, 1024); ASYNC_B128(ldsa, ga, 1536);
  ASYNC_B128(ldsa, ga, 2048); ASYNC_B128(ldsa, ga, 2560);
  ASYNC_B128(ldsa, ga, 3072); ASYNC_B128(ldsa, ga, 3584);
  ASYNC_B128(ldsa, ga, 4096); ASYNC_B128(ldsa, ga, 4608);
  ASYNC_B128(ldsa, ga, 5120); ASYNC_B128(ldsa, ga, 5632);
  ASYNC_B128(ldsa, ga, 6144); ASYNC_B128(ldsa, ga, 6656);
  ASYNC_B128(ldsa, ga, 7168); ASYNC_B128(ldsa, ga, 7680);
  S_WAIT_ASYNC0;
  __syncthreads();

  int m   = lane & 15;
  int kbA = (lane & 16) ? 8 : 0;
  v8f acc = {};
#pragma unroll
  for (int k0 = 0; k0 < PAIR_D; k0 += 32) {
    v16bf a = frag_A_f32(sAf + m * PAIR_D + k0 + kbA);
    v16bf b = load_B_ldsT(sBt, PAIR_D, k0, lane);
    acc = wmma_bf16(a, b, acc);
  }
  int h = lane & 15;
  if (h < H_NUM) {
#pragma unroll
    for (int r = 0; r < 8; ++r) {
      int pair = pair0 + ((lane >> 4) << 3) + r;
      int q = pair >> 9, kk = pair & (N_RES - 1);
      float mterm = -1e5f * (1.f - mask[q] * mask[kk]);
      logits[(size_t)pair * H_NUM + h] = acc[r] + bb[h] + mterm;
    }
  }
}

// ---------------------------------------------------------------------------
// Scalar-q.k + point cross terms, one WMMA per head per 16x16 (q,k) tile.
// Per-head feature vector (K=32): [16 scalar dims, 12 point dims, 4 zeros].
// w_h folded into Q-side point dims; -0.5*w_h*|pt|^2 norms added in epilogue.
// ---------------------------------------------------------------------------
__global__ __launch_bounds__(32) void qk_logits_kernel(
    const float* __restrict__ q_s, const float* __restrict__ k_s,
    const float* __restrict__ q_pt, const float* __restrict__ k_pt,
    const float* __restrict__ tpw, float* __restrict__ logits) {
  __shared__ __bf16 sA[16 * 32];
  __shared__ __bf16 sBt[16 * 32];   // transposed [n][k]
  __shared__ float sQN[16], sKN[16];
  int lane = threadIdx.x;
  int q0 = blockIdx.x * 16, k0 = blockIdx.y * 16;
  const float sw = 0.25f;                 // sqrt(1/SQK)
  const int HPQ = H_NUM * PQK_D;          // 48
  for (int h = 0; h < H_NUM; ++h) {
    float wh = 0.23570226f * log1pf(__expf(tpw[h]));  // sqrt(1/18)*softplus
    int r  = lane >> 1;
    int cb = (lane & 1) << 4;
#pragma unroll
    for (int i = 0; i < 16; ++i) {
      int col = cb + i;
      float v;
      if (col < 16)      v = q_s[(size_t)(q0 + r) * (H_NUM * SQK_D) + h * SQK_D + col] * sw;
      else if (col < 28) { int d = col - 16; int c3 = d >> 2, p = d & 3;
        v = q_pt[((size_t)(q0 + r) * 3 + c3) * HPQ + h * PQK_D + p] * wh; }
      else v = 0.f;
      sA[r * 32 + col] = (__bf16)v;
    }
#pragma unroll
    for (int i = 0; i < 16; ++i) {
      int d = lane;  // feature index 0..31
      float v;
      if (d < 16)      v = k_s[(size_t)(k0 + i) * (H_NUM * SQK_D) + h * SQK_D + d];
      else if (d < 28) { int dd = d - 16; int c3 = dd >> 2, p = dd & 3;
        v = k_pt[((size_t)(k0 + i) * 3 + c3) * HPQ + h * PQK_D + p]; }
      else v = 0.f;
      sBt[i * 32 + d] = (__bf16)v;
    }
    {  // point-norm rank-1 terms: lanes 0-15 -> Q side, 16-31 -> K side
      int idx = lane & 15;
      const float* pt = (lane < 16) ? q_pt : k_pt;
      int nn = ((lane < 16) ? q0 : k0) + idx;
      float s = 0.f;
#pragma unroll
      for (int c3 = 0; c3 < 3; ++c3)
#pragma unroll
        for (int p = 0; p < PQK_D; ++p) {
          float x = pt[((size_t)nn * 3 + c3) * HPQ + h * PQK_D + p];
          s += x * x;
        }
      if (lane < 16) sQN[idx] = -0.5f * wh * s; else sKN[idx] = -0.5f * wh * s;
    }
    __syncthreads();
    v8f acc = {};
    acc = wmma_bf16(load_A_lds(sA, 32, lane), load_B_ldsT(sBt, 32, 0, lane), acc);
    int nn = lane & 15;
    int mb = (lane >> 4) << 3;
#pragma unroll
    for (int rr = 0; rr < 8; ++rr) {
      size_t off = ((size_t)(q0 + mb + rr) * N_RES + (k0 + nn)) * H_NUM + h;
      logits[off] += acc[rr] + sQN[mb + rr] + sKN[nn];
    }
    __syncthreads();
  }
}

// ---------------------------------------------------------------------------
// Softmax over keys per (q,h); writes attn transposed [h][q][k] (contig k).
// ---------------------------------------------------------------------------
__global__ __launch_bounds__(32) void softmax_kernel(
    const float* __restrict__ logits, float* __restrict__ attn_t) {
  int q = blockIdx.x, h = blockIdx.y;
  int lane = threadIdx.x;
  const float scale = 0.57735026919f;     // sqrt(1/3)
  float vals[16];
  float mx = -3.4e38f;
#pragma unroll
  for (int i = 0; i < 16; ++i) {
    int k = lane + i * 32;
    float v = logits[((size_t)q * N_RES + k) * H_NUM + h] * scale;
    vals[i] = v;
    mx = fmaxf(mx, v);
  }
  for (int off = 16; off; off >>= 1) mx = fmaxf(mx, __shfl_xor(mx, off, 32));
  float sum = 0.f;
#pragma unroll
  for (int i = 0; i < 16; ++i) { vals[i] = __expf(vals[i] - mx); sum += vals[i]; }
  for (int off = 16; off; off >>= 1) sum += __shfl_xor(sum, off, 32);
  float inv = 1.f / sum;
#pragma unroll
  for (int i = 0; i < 16; ++i)
    attn_t[((size_t)h * N_RES + q) * N_RES + lane + i * 32] = vals[i] * inv;
}

// ---------------------------------------------------------------------------
// attn @ [v_s | v_pt] per head: M=512(q), K=512(k), N=40 (16 scalar + 3*8 pt).
// Scalar part -> final_act[:,0:192]; point part -> res_pt (N,3,H*PV).
// ---------------------------------------------------------------------------
__global__ __launch_bounds__(32) void pav_kernel(
    const float* __restrict__ attn_t, const float* __restrict__ v_s,
    const float* __restrict__ v_pt, float* __restrict__ final_act,
    float* __restrict__ res_pt) {
  int lane = threadIdx.x;
  int q0 = blockIdx.x * 16, n0 = blockIdx.y * 16, h = blockIdx.z;
  int m   = lane & 15;
  int kbA = (lane & 16) ? 8 : 0;
  int kbB = (lane & 16) ? 8 : 0;
  int nn  = lane & 15;
  int cc  = n0 + nn;
  const int HPV = H_NUM * PV_D;  // 96
  const float* arow = attn_t + ((size_t)h * N_RES + q0 + m) * N_RES;
  v8f acc = {};
  for (int k0 = 0; k0 < N_RES; k0 += 32) {
    v16bf a = frag_A_f32(arow + k0 + kbA);
    v16bf b;
#pragma unroll
    for (int e = 0; e < 16; ++e) {
      int k = k0 + kbB + e + (e & 8);
      float v;
      if (cc < 16)      v = v_s[(size_t)k * (H_NUM * SV_D) + h * SV_D + cc];
      else if (cc < 40) { int d = cc - 16; int c3 = d >> 3, p = d & 7;
        v = v_pt[((size_t)k * 3 + c3) * HPV + h * PV_D + p]; }
      else v = 0.f;
      b[e] = (__bf16)v;
    }
    acc = wmma_bf16(a, b, acc);
  }
  if (cc < 40) {
#pragma unroll
    for (int r = 0; r < 8; ++r) {
      int q = q0 + ((lane >> 4) << 3) + r;
      if (cc < 16)
        final_act[(size_t)q * ACT_DIM + h * SV_D + cc] = acc[r];
      else {
        int d = cc - 16; int c3 = d >> 3, p = d & 7;
        res_pt[((size_t)q * 3 + c3) * HPV + h * PV_D + p] = acc[r];
      }
    }
  }
}

// ---------------------------------------------------------------------------
// Invert frames (R^T * (g - t)), point norms -> final_act[:,192:576].
// ---------------------------------------------------------------------------
__global__ void invert_kernel(const float* __restrict__ res_pt,
                              const float* __restrict__ rot,
                              const float* __restrict__ trans,
                              float* __restrict__ final_act) {
  int n = blockIdx.x;
  int hp = threadIdx.x;
  const int HPV = H_NUM * PV_D;
  if (hp >= HPV) return;
  const float* R = rot + n * 9;
  float g0 = res_pt[((size_t)n * 3 + 0) * HPV + hp] - trans[n * 3 + 0];
  float g1 = res_pt[((size_t)n * 3 + 1) * HPV + hp] - trans[n * 3 + 1];
  float g2 = res_pt[((size_t)n * 3 + 2) * HPV + hp] - trans[n * 3 + 2];
  float l0 = R[0] * g0 + R[3] * g1 + R[6] * g2;
  float l1 = R[1] * g0 + R[4] * g1 + R[7] * g2;
  float l2 = R[2] * g0 + R[5] * g1 + R[8] * g2;
  size_t base = (size_t)n * ACT_DIM;
  final_act[base + 192 + hp] = l0;
  final_act[base + 288 + hp] = l1;
  final_act[base + 384 + hp] = l2;
  final_act[base + 480 + hp] = sqrtf(l0 * l0 + l1 * l1 + l2 * l2 + 1e-8f);
}

// ---------------------------------------------------------------------------
// attn_over_2d per q: A[h,k]=attn (12 rows + 4 zero pad), B=inputs_2d[q].
// Second (L2-resident) pass over inputs_2d; the 32x16 f32 B tile is staged
// with async b128 copies (global col-chunk stride 16B == LDS stride 16B).
// -> final_act[:,576:2112].
// ---------------------------------------------------------------------------
__global__ __launch_bounds__(32) void attn2d_kernel(
    const float* __restrict__ attn_t, const float* __restrict__ in2d,
    float* __restrict__ final_act) {
  __shared__ float sBf[32 * 16];   // [k][n] f32, 2KB
  int lane = threadIdx.x;
  int n0 = blockIdx.x * 16;
  int q  = blockIdx.y;
  int hrow = lane & 15;
  int kbA = (lane & 16) ? 8 : 0;
  int kbB = (lane & 16) ? 8 : 0;
  int nn  = lane & 15;
  const float* arow =
      (hrow < H_NUM) ? attn_t + ((size_t)hrow * N_RES + q) * N_RES : nullptr;
  const float* brow = in2d + (size_t)q * N_RES * PAIR_D;
  unsigned ldsb = (unsigned)(size_t)sBf + (unsigned)lane * 64u;
  v8f acc = {};
  for (int k0 = 0; k0 < N_RES; k0 += 32) {
    // async-stage B: row k0+lane, cols n0..n0+15 (64B) in 4 x b128
    unsigned long long gb = (unsigned long long)(size_t)(
        brow + (size_t)(k0 + lane) * PAIR_D + n0);
    ASYNC_B128(ldsb, gb, 0);
    ASYNC_B128(ldsb, gb, 16);
    ASYNC_B128(ldsb, gb, 32);
    ASYNC_B128(ldsb, gb, 48);
    v16bf a;
    if (arow) a = frag_A_f32(arow + k0 + kbA);
    else {
      v16bf z = {};
      a = z;
    }
    S_WAIT_ASYNC0;
    __syncthreads();
    v16bf b;
#pragma unroll
    for (int e = 0; e < 16; ++e)
      b[e] = (__bf16)sBf[(kbB + e + (e & 8)) * 16 + nn];
    acc = wmma_bf16(a, b, acc);
    S_WAIT_DS0;   // frag reads done before next iteration's async overwrite
    __syncthreads();
  }
#pragma unroll
  for (int r = 0; r < 8; ++r) {
    int h = ((lane >> 4) << 3) + r;
    if (h < H_NUM)
      final_act[(size_t)q * ACT_DIM + 576 + h * PAIR_D + n0 + nn] = acc[r];
  }
}

// ---------------------------------------------------------------------------
extern "C" void kernel_launch(void* const* d_in, const int* in_sizes, int n_in,
                              void* d_out, int out_size, void* d_ws, size_t ws_size,
                              hipStream_t stream) {
  (void)in_sizes; (void)n_in; (void)out_size; (void)ws_size;
  const float* in1d  = (const float*)d_in[0];
  const float* in2d  = (const float*)d_in[1];
  const float* mask  = (const float*)d_in[2];
  const float* rot   = (const float*)d_in[3];
  const float* trans = (const float*)d_in[4];
  const float* Wqs   = (const float*)d_in[5];
  const float* Wks   = (const float*)d_in[6];
  const float* Wvs   = (const float*)d_in[7];
  const float* Wqp   = (const float*)d_in[8];
  const float* bqp   = (const float*)d_in[9];
  const float* Wkp   = (const float*)d_in[10];
  const float* bkp   = (const float*)d_in[11];
  const float* Wvp   = (const float*)d_in[12];
  const float* bvp   = (const float*)d_in[13];
  const float* tpw   = (const float*)d_in[14];
  const float* W2d   = (const float*)d_in[15];
  const float* b2d   = (const float*)d_in[16];
  const float* Wout  = (const float*)d_in[17];
  const float* bout  = (const float*)d_in[18];
  float* out = (float*)d_out;

  float* ws = (float*)d_ws;
  float* q_s    = ws; ws += N_RES * H_NUM * SQK_D;        // 512*192
  float* k_s    = ws; ws += N_RES * H_NUM * SQK_D;
  float* v_s    = ws; ws += N_RES * H_NUM * SV_D;
  float* qp_raw = ws; ws += N_RES * H_NUM * 3 * PQK_D;    // 512*144
  float* kp_raw = ws; ws += N_RES * H_NUM * 3 * PQK_D;
  float* vp_raw = ws; ws += N_RES * H_NUM * 3 * PV_D;     // 512*288
  float* q_pt   = ws; ws += N_RES * 3 * H_NUM * PQK_D;
  float* k_pt   = ws; ws += N_RES * 3 * H_NUM * PQK_D;
  float* v_pt   = ws; ws += N_RES * 3 * H_NUM * PV_D;
  float* logits = ws; ws += (size_t)N_RES * N_RES * H_NUM; // 3.1M f32
  float* attn_t = ws; ws += (size_t)N_RES * N_RES * H_NUM;
  float* res_pt = ws; ws += N_RES * 3 * H_NUM * PV_D;
  float* f_act  = ws; ws += (size_t)N_RES * ACT_DIM;

  // 1) projections (WMMA GEMMs)
  gemm_bf16_kernel<<<dim3(32, 12), 32, 0, stream>>>(in1d, Wqs, nullptr, q_s, N_RES, 192, C_DIM);
  gemm_bf16_kernel<<<dim3(32, 12), 32, 0, stream>>>(in1d, Wks, nullptr, k_s, N_RES, 192, C_DIM);
  gemm_bf16_kernel<<<dim3(32, 12), 32, 0, stream>>>(in1d, Wvs, nullptr, v_s, N_RES, 192, C_DIM);
  gemm_bf16_kernel<<<dim3(32,  9), 32, 0, stream>>>(in1d, Wqp, bqp, qp_raw, N_RES, 144, C_DIM);
  gemm_bf16_kernel<<<dim3(32,  9), 32, 0, stream>>>(in1d, Wkp, bkp, kp_raw, N_RES, 144, C_DIM);
  gemm_bf16_kernel<<<dim3(32, 18), 32, 0, stream>>>(in1d, Wvp, bvp, vp_raw, N_RES, 288, C_DIM);

  // 2) points -> global frame
  to_global_kernel<<<N_RES, 160, 0, stream>>>(qp_raw, rot, trans, q_pt, PQK_D);
  to_global_kernel<<<N_RES, 160, 0, stream>>>(kp_raw, rot, trans, k_pt, PQK_D);
  to_global_kernel<<<N_RES, 288, 0, stream>>>(vp_raw, rot, trans, v_pt, PV_D);

  // 3) pair bias + mask into logits (first streaming pass over inputs_2d)
  bias_logits_kernel<<<(N_RES * N_RES) / 16, 32, 0, stream>>>(in2d, W2d, b2d, mask, logits);

  // 4) scalar + point qk terms
  qk_logits_kernel<<<dim3(32, 32), 32, 0, stream>>>(q_s, k_s, q_pt, k_pt, tpw, logits);

  // 5) softmax over keys -> attn transposed [h][q][k]
  softmax_kernel<<<dim3(N_RES, H_NUM), 32, 0, stream>>>(logits, attn_t);

  // 6) attn @ [v_s | v_pt]
  pav_kernel<<<dim3(32, 3, H_NUM), 32, 0, stream>>>(attn_t, v_s, v_pt, f_act, res_pt);

  // 7) invert frames + point norms
  invert_kernel<<<N_RES, 96, 0, stream>>>(res_pt, rot, trans, f_act);

  // 8) attn_over_2d (second pass over inputs_2d, L2-resident)
  attn2d_kernel<<<dim3(8, N_RES), 32, 0, stream>>>(attn_t, in2d, f_act);

  // 9) final projection (K = 2112 = 66*32)
  gemm_bf16_kernel<<<dim3(32, 24), 32, 0, stream>>>(f_act, Wout, bout, out,
                                                    N_RES, C_DIM, ACT_DIM);
}
